// EventSpikeAutoencoder_91036126806492
// MI455X (gfx1250) — matrix operand order repaired and checked by reading, not compiled, not run on hardware
//
#include <hip/hip_runtime.h>

// ---- problem sizes -------------------------------------------------------
#define SB 256           // batch
#define ST 512           // time steps
#define SD 128           // input dim
#define SH 256           // hidden dim
#define SE 64            // embed dim
#define SN (SB*ST)       // flattened rows

typedef __attribute__((ext_vector_type(16))) __bf16 v16bf;
typedef __attribute__((ext_vector_type(8)))  float  v8f;
typedef int vi4 __attribute__((__vector_size__(4 * sizeof(int))));

__device__ __forceinline__ v8f v8f_zero(){
  v8f z;
#pragma unroll
  for (int i = 0; i < 8; ++i) z[i] = 0.0f;
  return z;
}

__device__ __forceinline__ float gelu_exact(float x){
  return 0.5f * x * (1.0f + erff(x * 0.70710678118654752f));
}

// ---- gfx1250 async global->LDS staging (ASYNCcnt path) -------------------
#if defined(__gfx1250__) && __has_builtin(__builtin_amdgcn_global_load_async_to_lds_b128)
#define HAVE_ASYNC_LDS 1
#else
#define HAVE_ASYNC_LDS 0
#endif

__device__ __forceinline__ void async_copy16(void* lds_dst, const void* gsrc){
#if HAVE_ASYNC_LDS
  __builtin_amdgcn_global_load_async_to_lds_b128(
      (__attribute__((address_space(1))) vi4*)gsrc,
      (__attribute__((address_space(3))) vi4*)lds_dst, 0, 0);
#else
  *(float4*)lds_dst = *(const float4*)gsrc;
#endif
}
__device__ __forceinline__ void async_wait(){
#if HAVE_ASYNC_LDS
  asm volatile("s_wait_asynccnt 0" ::: "memory");
#endif
}

// ---- WMMA bf16 fragment loaders (CDNA5 ISA 7.12.2 layouts) ---------------
// A 16x32 (MxK), 16-bit: lane l (m=l&15, kh=l>>4):
//   elem e<8  -> K = kc + kh*8 + e
//   elem e>=8 -> K = kc + 16 + kh*8 + (e-8)
// Both 8-element groups are contiguous bf16 -> two 16B LDS loads.
__device__ __forceinline__ v16bf frag_a_bf16(const __bf16* A, int lda, int kc, int lane){
  const int m = lane & 15, kh = lane >> 4;
  const __bf16* p = A + m*lda + kc + kh*8;
  v16bf a;
#pragma unroll
  for (int e = 0; e < 8; ++e){ a[e] = p[e]; a[e+8] = p[e+16]; }
  return a;
}
// B 32x16 (KxN), 16-bit, read from a column-major LDS image BT[col][K]:
// lane l (n=l&15, kh=l>>4): elem e<8 -> K = kc + kh*8 + e, e>=8 -> K = kc + 16 + kh*8 + (e-8)
__device__ __forceinline__ v16bf frag_b_bf16(const __bf16* BT, int ldb, int kc, int c0, int lane){
  const int n = lane & 15, kh = lane >> 4;
  const __bf16* p = BT + (size_t)(c0 + n)*ldb + kc + kh*8;
  v16bf b;
#pragma unroll
  for (int e = 0; e < 8; ++e){ b[e] = p[e]; b[e+8] = p[e+16]; }
  return b;
}

// ==========================================================================
// Kernel 1: cur_in = LN(x) @ W_in + mask @ W_mask + b_in        [SN x SH]
// Tile: 64 rows x 128 cols per workgroup (256 thr = 8 waves, 4 tiles/wave).
// ==========================================================================
__global__ __launch_bounds__(256) void k_curin(
    const float* __restrict__ x,    const float* __restrict__ msk,
    const float* __restrict__ ln_g, const float* __restrict__ ln_b,
    const float* __restrict__ W_in, const float* __restrict__ b_in,
    const float* __restrict__ W_mask, float* __restrict__ curin)
{
  __shared__ __align__(16) float  Xs[64*SD];     // f32 staging (LN in f32)
  __shared__ __align__(16) __bf16 As[64*136];    // bf16 A tile, K padded to 136
  __shared__ __align__(16) __bf16 BT[128*136];   // bf16 B tile, transposed [col][K]
  const int tid = threadIdx.x, lane = tid & 31, wave = tid >> 5;
  const int n0 = blockIdx.x * 64;
  const int c0 = blockIdx.y * 128;
  const int rt = wave & 3;                       // row-tile (16 rows)
  const int cb = 64 * (wave >> 2);               // col base within 128

  v8f acc[4];
#pragma unroll
  for (int i = 0; i < 4; ++i) acc[i] = v8f_zero();

  for (int phase = 0; phase < 2; ++phase){
    const float* Asrc = phase ? msk    : x;
    const float* Bsrc = phase ? W_mask : W_in;
    for (int i = tid; i < 64*SD/4; i += 256)     // async tile stage, 16B/lane
      async_copy16(Xs + i*4, Asrc + (size_t)n0*SD + (size_t)i*4);
    async_wait();
    __syncthreads();
    if (phase == 0 && tid < 64){                 // layernorm one row per thread
      float mu = 0.0f;
      for (int k = 0; k < SD; ++k) mu += Xs[tid*SD + k];
      mu *= (1.0f/SD);
      float var = 0.0f;
      for (int k = 0; k < SD; ++k){ float d = Xs[tid*SD + k] - mu; var += d*d; }
      float rs = rsqrtf(var*(1.0f/SD) + 1e-5f);
      for (int k = 0; k < SD; ++k)
        Xs[tid*SD + k] = (Xs[tid*SD + k] - mu)*rs*ln_g[k] + ln_b[k];
    }
    __syncthreads();
    for (int i = tid; i < 64*SD; i += 256){ int r = i >> 7, k = i & 127; As[r*136 + k] = (__bf16)Xs[i]; }
    for (int i = tid; i < 128*SD; i += 256){
      int k = i >> 7, c = i & 127;               // coalesced read, transposed write
      BT[c*136 + k] = (__bf16)Bsrc[(size_t)k*SH + c0 + c];
    }
    __syncthreads();
    for (int kc = 0; kc < SD; kc += 32){
      v16bf a = frag_a_bf16(As + 16*rt*136, 136, kc, lane);
#pragma unroll
      for (int ct = 0; ct < 4; ++ct){
        v16bf b = frag_b_bf16(BT, 136, kc, cb + 16*ct, lane);
        acc[ct] = __builtin_amdgcn_wmma_f32_16x16x32_bf16(false, a, false, b, (short)0, acc[ct], false, false);
      }
    }
    __syncthreads();
  }
  const int n_ = lane & 15, kh = lane >> 4;
  for (int ct = 0; ct < 4; ++ct){
    const int col = c0 + cb + 16*ct + n_;
    const float bb = b_in[col];
#pragma unroll
    for (int r = 0; r < 8; ++r){
      const int row = n0 + 16*rt + r + 8*kh;     // C/D layout: M = r + 8*kh
      curin[(size_t)row*SH + col] = acc[ct][r] + bb;
    }
  }
}

// ==========================================================================
// Kernel 2: LIF recurrent scan. One WG owns 16 batch rows for all T steps.
// W_rec lives transposed in LDS as bf16 (132 KB, CDNA5 320KB LDS).
// Membrane state stays in WMMA accumulator registers; spikes shared via LDS.
// ==========================================================================
__global__ __launch_bounds__(256) void k_scan(
    const float* __restrict__ curin, const float* __restrict__ W_rec,
    float* __restrict__ spk, float* __restrict__ memt, float* __restrict__ pooled)
{
  __shared__ __align__(16) __bf16 WrecT[SH*264];   // [col][K], K padded to 264
  __shared__ __align__(16) __bf16 Sp[16*264];      // current spikes [m][K]
  const int tid = threadIdx.x, lane = tid & 31, wave = tid >> 5;
  const int b0 = blockIdx.x * 16;
  const int c0 = wave * 32;                        // this wave's 32 hidden cols

  for (int i = tid; i < SH*SH; i += 256){          // convert + transpose W_rec
    int k = i >> 8, c = i & 255;
    WrecT[c*264 + k] = (__bf16)W_rec[i];
  }
  for (int i = tid; i < 16*264; i += 256) Sp[i] = (__bf16)0.0f;
  __syncthreads();

  v8f mem[2], ssum[2];
  mem[0] = v8f_zero(); mem[1] = v8f_zero();
  ssum[0] = v8f_zero(); ssum[1] = v8f_zero();
  const int n_ = lane & 15, kh = lane >> 4;

  for (int t = 0; t < ST; ++t){
    v8f rec[2]; rec[0] = v8f_zero(); rec[1] = v8f_zero();
#pragma unroll
    for (int kc = 0; kc < SH; kc += 32){
      v16bf a = frag_a_bf16(Sp, 264, kc, lane);
#pragma unroll
      for (int ct = 0; ct < 2; ++ct){
        v16bf b = frag_b_bf16(WrecT, 264, kc, c0 + 16*ct, lane);
        rec[ct] = __builtin_amdgcn_wmma_f32_16x16x32_bf16(false, a, false, b, (short)0, rec[ct], false, false);
      }
    }
    __syncthreads();                               // everyone done reading Sp
#pragma unroll
    for (int ct = 0; ct < 2; ++ct){
      const int col = c0 + 16*ct + n_;
#pragma unroll
      for (int r = 0; r < 8; ++r){
        const int m = r + 8*kh;
        const size_t gi = (size_t)(b0 + m)*((size_t)ST*SH) + (size_t)t*SH + col;
        if (r == 0 && t + 1 < ST) __builtin_prefetch(curin + gi + SH, 0, 1);
        float cur = curin[gi] + 0.25f * rec[ct][r];        // RSCALE
        float mv  = 0.9f * mem[ct][r] + cur;               // BETA
        float s   = (mv - 1.0f >= 0.0f) ? 1.0f : 0.0f;     // THR
        mv -= s;                                           // s * THR
        mem[ct][r]  = mv;
        ssum[ct][r] += s;
        spk[gi]  = s;
        memt[gi] = mv;
        Sp[m*264 + col] = (__bf16)s;                       // exact in bf16
      }
    }
    __syncthreads();                               // new spikes visible
  }
#pragma unroll
  for (int ct = 0; ct < 2; ++ct){
    const int col = c0 + 16*ct + n_;
#pragma unroll
    for (int r = 0; r < 8; ++r){
      const int m = r + 8*kh;
      pooled[(size_t)(b0+m)*(2*SH) + col]      = ssum[ct][r] * (1.0f/ST);
      pooled[(size_t)(b0+m)*(2*SH) + SH + col] = mem[ct][r];
    }
  }
}

// ==========================================================================
// Kernel 3: fused decoder per 32-row tile:
//   LN2([spike|mem]) @ Wd1 + bd1 -> gelu -> @ Wd2 + bd2 -> reconstruction
// ==========================================================================
__global__ __launch_bounds__(256) void k_decoder(
    const float* __restrict__ spk, const float* __restrict__ memt,
    const float* __restrict__ ln2_g, const float* __restrict__ ln2_b,
    const float* __restrict__ Wd1, const float* __restrict__ bd1,
    const float* __restrict__ Wd2, const float* __restrict__ bd2,
    float* __restrict__ out)
{
  __shared__ __align__(16) char smem[69632 + 33280 + 16896];
  __shared__ float rmu[32], rrs[32];
  float*  Fs = (float*)smem;                       // [32][512] f32 staging (aliases BT)
  __bf16* BT = (__bf16*)smem;                      // weight image [col][K]
  __bf16* As = (__bf16*)(smem + 69632);            // [32][520] LN'd traces bf16
  __bf16* Hs = (__bf16*)(smem + 69632 + 33280);    // [32][264] gelu(h) bf16
  const int tid = threadIdx.x, lane = tid & 31, wave = tid >> 5;
  const int n0 = blockIdx.x * 32;
  const int rt = wave & 1;

  // stage trace rows [spike | membrane] via async global->LDS, 16B chunks
  for (int i = tid; i < 32*128; i += 256){         // i in float4 units
    int r = i >> 7, c4 = i & 127;
    const float* g = (c4 < 64) ? (spk  + (size_t)(n0+r)*SH + (size_t)c4*4)
                               : (memt + (size_t)(n0+r)*SH + (size_t)(c4-64)*4);
    async_copy16(Fs + (size_t)i*4, g);
  }
  async_wait();
  __syncthreads();
  if (tid < 32){                                   // LN2 stats in f32
    float mu = 0.0f;
    for (int k = 0; k < 512; ++k) mu += Fs[tid*512 + k];
    mu *= (1.0f/512.0f);
    float var = 0.0f;
    for (int k = 0; k < 512; ++k){ float d = Fs[tid*512 + k] - mu; var += d*d; }
    rmu[tid] = mu; rrs[tid] = rsqrtf(var*(1.0f/512.0f) + 1e-5f);
  }
  __syncthreads();
  for (int i = tid; i < 32*512; i += 256){
    int r = i >> 9, c = i & 511;
    As[r*520 + c] = (__bf16)((Fs[i] - rmu[r])*rrs[r]*ln2_g[c] + ln2_b[c]);
  }
  __syncthreads();                                 // Fs dead; BT may reuse region

  // phase 1: [32x512] @ Wd1[512x256], K-chunks of 128
  v8f acc[4];
#pragma unroll
  for (int i = 0; i < 4; ++i) acc[i] = v8f_zero();
  const int cb1 = 64 * (wave >> 1);
  for (int kc = 0; kc < 512; kc += 128){
    for (int i = tid; i < 128*SH; i += 256){
      int k = i >> 8, c = i & 255;
      BT[c*136 + k] = (__bf16)Wd1[(size_t)(kc + k)*SH + c];
    }
    __syncthreads();
#pragma unroll
    for (int k0 = 0; k0 < 128; k0 += 32){
      v16bf a = frag_a_bf16(As + 16*rt*520, 520, kc + k0, lane);
#pragma unroll
      for (int ct = 0; ct < 4; ++ct){
        v16bf b = frag_b_bf16(BT, 136, k0, cb1 + 16*ct, lane);
        acc[ct] = __builtin_amdgcn_wmma_f32_16x16x32_bf16(false, a, false, b, (short)0, acc[ct], false, false);
      }
    }
    __syncthreads();
  }
  const int n_ = lane & 15, kh = lane >> 4;
  for (int ct = 0; ct < 4; ++ct){
    const int col = cb1 + 16*ct + n_;
    const float bb = bd1[col];
#pragma unroll
    for (int r = 0; r < 8; ++r){
      const int row = 16*rt + r + 8*kh;
      Hs[row*264 + col] = (__bf16)gelu_exact(acc[ct][r] + bb);
    }
  }
  __syncthreads();

  // phase 2: gelu(h)[32x256] @ Wd2[256x128]
  for (int i = tid; i < SH*SD; i += 256){
    int k = i >> 7, c = i & 127;
    BT[c*264 + k] = (__bf16)Wd2[(size_t)k*SD + c];
  }
  __syncthreads();
  v8f a2[2]; a2[0] = v8f_zero(); a2[1] = v8f_zero();
  const int cb2 = 32 * (wave >> 1);
#pragma unroll
  for (int kc = 0; kc < SH; kc += 32){
    v16bf a = frag_a_bf16(Hs + 16*rt*264, 264, kc, lane);
#pragma unroll
    for (int ct = 0; ct < 2; ++ct){
      v16bf b = frag_b_bf16(BT, 264, kc, cb2 + 16*ct, lane);
      a2[ct] = __builtin_amdgcn_wmma_f32_16x16x32_bf16(false, a, false, b, (short)0, a2[ct], false, false);
    }
  }
  for (int ct = 0; ct < 2; ++ct){
    const int col = cb2 + 16*ct + n_;
    const float bb = bd2[col];
#pragma unroll
    for (int r = 0; r < 8; ++r){
      const int row = n0 + 16*rt + r + 8*kh;
      out[(size_t)row*SD + col] = a2[ct][r] + bb;
    }
  }
}

// ==========================================================================
// Kernel 4: embedding head (0.1% of FLOPs -> plain VALU). One WG per batch row.
// ==========================================================================
__global__ __launch_bounds__(256) void k_embed(
    const float* __restrict__ pooled,
    const float* __restrict__ ln3_g, const float* __restrict__ ln3_b,
    const float* __restrict__ We1, const float* __restrict__ be1,
    const float* __restrict__ We2, const float* __restrict__ be2,
    float* __restrict__ emb)
{
  __shared__ float row[2*SH];
  __shared__ float ev[SH];
  __shared__ float st[2];
  const int tid = threadIdx.x;
  const int b = blockIdx.x;
  for (int i = tid; i < 2*SH; i += 256) row[i] = pooled[(size_t)b*(2*SH) + i];
  __syncthreads();
  if (tid == 0){
    float mu = 0.0f;
    for (int k = 0; k < 512; ++k) mu += row[k];
    mu *= (1.0f/512.0f);
    float var = 0.0f;
    for (int k = 0; k < 512; ++k){ float d = row[k] - mu; var += d*d; }
    st[0] = mu; st[1] = rsqrtf(var*(1.0f/512.0f) + 1e-5f);
  }
  __syncthreads();
  const float mu = st[0], rs = st[1];
  for (int i = tid; i < 512; i += 256)
    row[i] = (row[i] - mu)*rs*ln3_g[i] + ln3_b[i];
  __syncthreads();
  {
    float a = be1[tid];
    for (int k = 0; k < 512; ++k) a += row[k]*We1[(size_t)k*SH + tid];
    ev[tid] = gelu_exact(a);
  }
  __syncthreads();
  if (tid < SE){
    float a = be2[tid];
    for (int k = 0; k < SH; ++k) a += ev[k]*We2[(size_t)k*SE + tid];
    emb[(size_t)b*SE + tid] = a;
  }
}

// ==========================================================================
extern "C" void kernel_launch(void* const* d_in, const int* in_sizes, int n_in,
                              void* d_out, int out_size, void* d_ws, size_t ws_size,
                              hipStream_t stream)
{
  (void)in_sizes; (void)n_in; (void)out_size; (void)ws_size;
  const float* x      = (const float*)d_in[0];
  const float* mask   = (const float*)d_in[1];
  const float* ln_g   = (const float*)d_in[2];
  const float* ln_b   = (const float*)d_in[3];
  const float* W_in   = (const float*)d_in[4];
  const float* b_in   = (const float*)d_in[5];
  const float* W_mask = (const float*)d_in[6];
  const float* W_rec  = (const float*)d_in[7];
  const float* ln2_g  = (const float*)d_in[8];
  const float* ln2_b  = (const float*)d_in[9];
  const float* Wd1    = (const float*)d_in[10];
  const float* bd1    = (const float*)d_in[11];
  const float* Wd2    = (const float*)d_in[12];
  const float* bd2    = (const float*)d_in[13];
  const float* ln3_g  = (const float*)d_in[14];
  const float* ln3_b  = (const float*)d_in[15];
  const float* We1    = (const float*)d_in[16];
  const float* be1    = (const float*)d_in[17];
  const float* We2    = (const float*)d_in[18];
  const float* be2    = (const float*)d_in[19];

  float* out    = (float*)d_out;                      // reconstruction [SN*SD] ++ embedding [SB*SE]
  float* ws     = (float*)d_ws;
  float* curin  = ws;                                 // SN*SH
  float* spkT   = curin + (size_t)SN*SH;              // SN*SH
  float* memT   = spkT  + (size_t)SN*SH;              // SN*SH
  float* pooled = memT  + (size_t)SN*SH;              // SB*2*SH

  k_curin  <<<dim3(SN/64, SH/128), 256, 0, stream>>>(x, mask, ln_g, ln_b, W_in, b_in, W_mask, curin);
  k_scan   <<<dim3(SB/16),          256, 0, stream>>>(curin, W_rec, spkT, memT, pooled);
  k_decoder<<<dim3(SN/32),          256, 0, stream>>>(spkT, memT, ln2_g, ln2_b, Wd1, bd1, Wd2, bd2, out);
  k_embed  <<<dim3(SB),             256, 0, stream>>>(pooled, ln3_g, ln3_b, We1, be1, We2, be2, out + (size_t)SN*SD);
}
